// LSTM_31954556682691
// MI455X (gfx1250) — matrix operand order repaired and checked by reading, not compiled, not run on hardware
//
#include <hip/hip_runtime.h>
#include <hip/hip_bf16.h>

typedef __attribute__((ext_vector_type(16))) _Float16 v16h;
typedef __attribute__((ext_vector_type(8)))  float    v8f;

#define B_SZ 256
#define T_SZ 1024
#define FIN_SZ 64
#define H_SZ 128
#define G_SZ (4 * H_SZ)            // 512 gate columns (i,f,g,o)
#define NCLS_SZ 2
#define ROWS_PER_BLK 16            // batch rows per workgroup (one WMMA M-tile)
#define THREADS 256                // 8 waves (wave32)
#define NWAVE (THREADS / 32)
#define NTILES (G_SZ / 16)         // 32 N-tiles of 16
#define TILES_PER_WAVE (NTILES / NWAVE) // 4
#define KB_HH (H_SZ / 32)          // 4 K-blocks for the recurrent GEMM

// ---------------------------------------------------------------------------
// Packed B-fragment layout (a lane's 16 halves contiguous -> ds_load_b128 x2)
// frag element m of lane in fragment (kb, nt):
//   k = kb*32 + (lane>>4)*16 + m,  n = nt*16 + (lane&15),  value = W[n][k]
// flat half index = ((kb*NTILES + nt)*32 + lane)*16 + m
// ---------------------------------------------------------------------------
__device__ __host__ __forceinline__ size_t frag_off(int kb, int nt, int lane) {
    return ((size_t)((kb * NTILES + nt) * 32 + lane)) * 16;
}

// A: 16x32 f16 (MxK), row-major LDS tile with row stride `ld`, K offset k0.
// lanes 0-15: M=lane, halves 0..15 -> K 0..15
// lanes 16-31: M=lane-16, halves 0..15 -> K 16..31  (K-contiguous runs of 16)
__device__ __forceinline__ v16h load_A_frag(const _Float16* lds, int ld, int k0, int lane) {
    const int row = lane & 15;
    const int khi = lane >> 4; // 0 or 1
    v16h a;
#pragma unroll
    for (int m = 0; m < 16; ++m) {
        const int k = 16 * khi + m;
        a[m] = lds[row * ld + k0 + k];
    }
    return a;
}

__device__ __forceinline__ float sigmoidf_fast(float x) {
    return 1.0f / (1.0f + __expf(-x));
}

// CDNA5 async global->LDS DMA (ASYNCcnt-tracked, no VGPR round trip).
// GVS mode: mem_addr = saddr + voffset ; LDS dest = ldsoff (per-lane VGPR).
__device__ __forceinline__ void async_ld16(unsigned ldsoff, unsigned voff,
                                           unsigned long long saddr) {
    asm volatile("global_load_async_to_lds_b128 %0, %1, %2"
                 :: "v"(ldsoff), "v"(voff), "s"(saddr) : "memory");
}

// ---------------------------------------------------------------------------
// x f32 -> f16 (one shot, so both LSTM layers can use async f16 staging)
// ---------------------------------------------------------------------------
__global__ __launch_bounds__(256)
void cvt_f32_to_f16_kernel(const float* __restrict__ src, _Float16* __restrict__ dst,
                           size_t n) {
    const size_t i = (size_t)blockIdx.x * blockDim.x + threadIdx.x;
    if (i < n) dst[i] = (_Float16)src[i];
}

// ---------------------------------------------------------------------------
// Weight pack: W[4H, D] f32 -> packed f16 B-fragments (layout above)
// ---------------------------------------------------------------------------
__global__ __launch_bounds__(256)
void pack_wfrag_kernel(const float* __restrict__ W, _Float16* __restrict__ Wp, int D) {
    const int i = blockIdx.x * blockDim.x + threadIdx.x;
    const int total = (D / 32) * NTILES * 32 * 16;
    if (i < total) {
        const int m    = i & 15;
        const int lane = (i >> 4) & 31;
        const int rest = i >> 9;
        const int nt   = rest % NTILES;
        const int kb   = rest / NTILES;
        const int k = kb * 32 + (lane >> 4) * 16 + m;
        const int n = nt * 16 + (lane & 15);
        Wp[i] = (_Float16)W[n * D + k];
    }
}

// ---------------------------------------------------------------------------
// Persistent fused LSTM layer. One workgroup owns 16 batch rows, loops over
// all T steps. W_hh^T fragments live in 128 VGPRs/lane for the entire scan;
// W_ih^T packed fragments are LDS-resident; x_t tiles are double-buffered via
// async global->LDS DMA so the fetch for step t+1 overlaps step t's WMMAs.
// Per step:  g[16,512] = x_t[16,D] @ Wih^T + h[16,128] @ Whh^T   (WMMA f16)
// then elementwise i,f,g,o gate math updates (h,c) in LDS and streams h out.
// ---------------------------------------------------------------------------
template <int D>
__global__ __launch_bounds__(THREADS)
void lstm_layer_kernel(const _Float16* __restrict__ xin,    // [B][T][D] f16
                       const _Float16* __restrict__ Wih_p,  // packed frags
                       const _Float16* __restrict__ Whh_p,  // packed frags
                       const float* __restrict__ bih,
                       const float* __restrict__ bhh,
                       _Float16* __restrict__ h_seq)        // [B][T][128] f16
{
    extern __shared__ char smem[];
    _Float16* sWih  = (_Float16*)smem;                        // (D/32)*NTILES*512 f16
    _Float16* sXt   = sWih + (D / 32) * NTILES * 512;         // 2 * 16*D f16 (double buffer)
    _Float16* sH    = sXt + 2 * ROWS_PER_BLK * D;             // 16*128 f16
    float*    sC    = (float*)(sH + ROWS_PER_BLK * H_SZ);     // 16*128 f32
    float*    sG    = sC + ROWS_PER_BLK * H_SZ;               // 16*512 f32
    float*    sBias = sG + ROWS_PER_BLK * G_SZ;               // 512 f32

    const int tid   = threadIdx.x;
    const int lane  = tid & 31;
    const int wave  = tid >> 5;
    const int brow0 = blockIdx.x * ROWS_PER_BLK;

    // one-time staging: W_ih frags + combined bias into LDS, zero h/c state
    for (int i = tid; i < (D / 32) * NTILES * 512; i += THREADS) sWih[i] = Wih_p[i];
    for (int i = tid; i < G_SZ; i += THREADS)                    sBias[i] = bih[i] + bhh[i];
    for (int i = tid; i < ROWS_PER_BLK * H_SZ; i += THREADS) {
        sH[i] = (_Float16)0.0f;
        sC[i] = 0.0f;
    }

    // hoist W_hh B-fragments for this wave's 4 N-tiles into registers
    // (4 kb x 4 tiles x 8 VGPRs = 128 VGPRs/lane, resident across all T steps)
    v16h whhB[KB_HH][TILES_PER_WAVE];
#pragma unroll
    for (int kb = 0; kb < KB_HH; ++kb)
#pragma unroll
        for (int i = 0; i < TILES_PER_WAVE; ++i)
            whhB[kb][i] = *(const v16h*)(Whh_p + frag_off(kb, wave * TILES_PER_WAVE + i, lane));

    // per-thread async DMA parameters: thread owns one 16B chunk of the x tile
    const int CHUNKS = (ROWS_PER_BLK * D) / 8;       // 16B chunks per tile
    const bool xactive = tid < CHUNKS;               // whole waves idle when D=64
    unsigned ldsoff[2] = {0, 0};
    unsigned voff = 0;
    if (xactive) {
        const int half0 = tid * 8;
        const int r = half0 / D, k = half0 % D;
        voff = (unsigned)((r * T_SZ * D + k) * 2);                       // per-lane byte offset
        ldsoff[0] = (unsigned)(size_t)(void*)(sXt + half0);              // LDS addr = low 32 bits
        ldsoff[1] = ldsoff[0] + (unsigned)(ROWS_PER_BLK * D * 2);
    }
    const unsigned long long xbase =
        (unsigned long long)(size_t)xin + (unsigned long long)brow0 * T_SZ * D * 2;

    // prime the pipeline: DMA x_0 into buffer 0
    if (xactive) async_ld16(ldsoff[0], voff, xbase);
    __syncthreads();

    for (int t = 0; t < T_SZ; ++t) {
        // kick DMA for x_{t+1} into the other buffer, then wait for x_t only
        if (t + 1 < T_SZ) {
            if (xactive) async_ld16(ldsoff[(t + 1) & 1], voff,
                                    xbase + (unsigned long long)(t + 1) * D * 2);
            asm volatile("s_wait_asynccnt 1" ::: "memory");
        } else {
            asm volatile("s_wait_asynccnt 0" ::: "memory");
        }
        __syncthreads();   // x_t tile visible to all waves

        const _Float16* sXtCur = sXt + (t & 1) * ROWS_PER_BLK * D;

        // gate pre-activations via WMMA: each wave owns 4 N-tiles of 16 cols
#pragma unroll
        for (int i = 0; i < TILES_PER_WAVE; ++i) {
            const int nt = wave * TILES_PER_WAVE + i;
            v8f acc = {};
#pragma unroll
            for (int k0 = 0; k0 < D; k0 += 32) {
                v16h a = load_A_frag(sXtCur, D, k0, lane);
                const v16h b = *(const v16h*)(sWih + frag_off(k0 >> 5, nt, lane));
                acc = __builtin_amdgcn_wmma_f32_16x16x32_f16(
                    false, a, false, b, (short)0, acc, false, false);
            }
#pragma unroll
            for (int kb = 0; kb < KB_HH; ++kb) {
                v16h a = load_A_frag(sH, H_SZ, kb * 32, lane);
                acc = __builtin_amdgcn_wmma_f32_16x16x32_f16(
                    false, a, false, whhB[kb][i], (short)0, acc, false, false);
            }
            // scatter D-matrix: VGPR r holds M = r + 8*(lane>=16), N = lane&15
            const int mhi = (lane >> 4) * 8;
            const int col = nt * 16 + (lane & 15);
#pragma unroll
            for (int r = 0; r < 8; ++r)
                sG[(mhi + r) * G_SZ + col] = acc[r];
        }
        __syncthreads();

        // elementwise LSTM cell: 16*128 states, 8 per thread
#pragma unroll
        for (int e = 0; e < (ROWS_PER_BLK * H_SZ) / THREADS; ++e) {
            const int flat = e * THREADS + tid;
            const int r = flat >> 7;
            const int c = flat & (H_SZ - 1);
            const float gi = sG[r * G_SZ + c           ] + sBias[c];
            const float gf = sG[r * G_SZ + c + H_SZ    ] + sBias[c + H_SZ];
            const float gg = sG[r * G_SZ + c + 2 * H_SZ] + sBias[c + 2 * H_SZ];
            const float go = sG[r * G_SZ + c + 3 * H_SZ] + sBias[c + 3 * H_SZ];
            const float iv = sigmoidf_fast(gi);
            const float fv = sigmoidf_fast(gf);
            const float gv = tanhf(gg);
            const float ov = sigmoidf_fast(go);
            const float cv = fv * sC[flat] + iv * gv;
            const float hv = ov * tanhf(cv);
            sC[flat] = cv;
            sH[flat] = (_Float16)hv;
            h_seq[(size_t)(brow0 + r) * T_SZ * H_SZ + (size_t)t * H_SZ + c] = (_Float16)hv;
        }
        __syncthreads();
    }
}

// ---------------------------------------------------------------------------
// Dropout + linear head: out[b, :] = drop(h2[b, T-1, :]) @ W_out^T + b_out
// ---------------------------------------------------------------------------
__global__ __launch_bounds__(H_SZ)
void classify_kernel(const _Float16* __restrict__ h2_seq,
                     const float* __restrict__ W_out,
                     const float* __restrict__ b_out,
                     float* __restrict__ out)
{
    __shared__ float s0[H_SZ];
    __shared__ float s1[H_SZ];
    const int b = blockIdx.x;
    const int h = threadIdx.x;

    float hv = (float)h2_seq[(size_t)b * T_SZ * H_SZ + (size_t)(T_SZ - 1) * H_SZ + h];

    // deterministic hash-based dropout (p=0.3), scale 1/(1-p)
    unsigned v = (unsigned)(b * H_SZ + h) * 2654435761u;
    v ^= v >> 16; v *= 2246822519u; v ^= v >> 13;
    const float u = (float)(v & 0xFFFFFFu) * (1.0f / 16777216.0f);
    hv = (u >= 0.3f) ? hv * (1.0f / 0.7f) : 0.0f;

    s0[h] = hv * W_out[0 * H_SZ + h];
    s1[h] = hv * W_out[1 * H_SZ + h];
    __syncthreads();
    for (int s = H_SZ / 2; s > 0; s >>= 1) {
        if (h < s) { s0[h] += s0[h + s]; s1[h] += s1[h + s]; }
        __syncthreads();
    }
    if (h == 0) {
        out[b * NCLS_SZ + 0] = s0[0] + b_out[0];
        out[b * NCLS_SZ + 1] = s1[0] + b_out[1];
    }
}

// ---------------------------------------------------------------------------
extern "C" void kernel_launch(void* const* d_in, const int* in_sizes, int n_in,
                              void* d_out, int out_size, void* d_ws, size_t ws_size,
                              hipStream_t stream)
{
    const float* x     = (const float*)d_in[0];
    const float* W_ih0 = (const float*)d_in[1];
    const float* W_hh0 = (const float*)d_in[2];
    const float* b_ih0 = (const float*)d_in[3];
    const float* b_hh0 = (const float*)d_in[4];
    const float* W_ih1 = (const float*)d_in[5];
    const float* W_hh1 = (const float*)d_in[6];
    const float* b_ih1 = (const float*)d_in[7];
    const float* b_hh1 = (const float*)d_in[8];
    const float* W_out = (const float*)d_in[9];
    const float* b_out = (const float*)d_in[10];

    // workspace: packed-fragment weights + f16 x + h-sequences
    char* ws = (char*)d_ws;
    _Float16* wp_ih0 = (_Float16*)ws;  ws += (size_t)FIN_SZ * G_SZ * sizeof(_Float16);
    _Float16* wp_hh0 = (_Float16*)ws;  ws += (size_t)H_SZ * G_SZ * sizeof(_Float16);
    _Float16* wp_ih1 = (_Float16*)ws;  ws += (size_t)H_SZ * G_SZ * sizeof(_Float16);
    _Float16* wp_hh1 = (_Float16*)ws;  ws += (size_t)H_SZ * G_SZ * sizeof(_Float16);
    _Float16* x16    = (_Float16*)ws;  ws += (size_t)B_SZ * T_SZ * FIN_SZ * sizeof(_Float16);
    _Float16* h1_seq = (_Float16*)ws;  ws += (size_t)B_SZ * T_SZ * H_SZ * sizeof(_Float16);
    _Float16* h2_seq = (_Float16*)ws;  ws += (size_t)B_SZ * T_SZ * H_SZ * sizeof(_Float16);

    const size_t nx = (size_t)B_SZ * T_SZ * FIN_SZ;
    cvt_f32_to_f16_kernel<<<dim3((unsigned)((nx + 255) / 256)), dim3(256), 0, stream>>>(x, x16, nx);

    pack_wfrag_kernel<<<dim3((G_SZ * FIN_SZ + 255) / 256), dim3(256), 0, stream>>>(W_ih0, wp_ih0, FIN_SZ);
    pack_wfrag_kernel<<<dim3((G_SZ * H_SZ  + 255) / 256), dim3(256), 0, stream>>>(W_hh0, wp_hh0, H_SZ);
    pack_wfrag_kernel<<<dim3((G_SZ * H_SZ  + 255) / 256), dim3(256), 0, stream>>>(W_ih1, wp_ih1, H_SZ);
    pack_wfrag_kernel<<<dim3((G_SZ * H_SZ  + 255) / 256), dim3(256), 0, stream>>>(W_hh1, wp_hh1, H_SZ);

    // dynamic LDS: Wih frags + 2x xt + h + c + g + bias  (W_hh lives in VGPRs)
    const size_t lds0 =
        ((size_t)FIN_SZ * G_SZ + 2 * ROWS_PER_BLK * FIN_SZ + ROWS_PER_BLK * H_SZ) * sizeof(_Float16) +
        ((size_t)ROWS_PER_BLK * H_SZ + (size_t)ROWS_PER_BLK * G_SZ + G_SZ) * sizeof(float);   // ~114 KB
    const size_t lds1 =
        ((size_t)H_SZ * G_SZ + 2 * ROWS_PER_BLK * H_SZ + ROWS_PER_BLK * H_SZ) * sizeof(_Float16) +
        ((size_t)ROWS_PER_BLK * H_SZ + (size_t)ROWS_PER_BLK * G_SZ + G_SZ) * sizeof(float);   // ~182 KB (fits 320 KB/WGP)

    (void)hipFuncSetAttribute(reinterpret_cast<const void*>(&lstm_layer_kernel<FIN_SZ>),
                              hipFuncAttributeMaxDynamicSharedMemorySize, (int)lds0);
    (void)hipFuncSetAttribute(reinterpret_cast<const void*>(&lstm_layer_kernel<H_SZ>),
                              hipFuncAttributeMaxDynamicSharedMemorySize, (int)lds1);

    lstm_layer_kernel<FIN_SZ><<<dim3(B_SZ / ROWS_PER_BLK), dim3(THREADS), lds0, stream>>>(
        x16, wp_ih0, wp_hh0, b_ih0, b_hh0, h1_seq);

    lstm_layer_kernel<H_SZ><<<dim3(B_SZ / ROWS_PER_BLK), dim3(THREADS), lds1, stream>>>(
        h1_seq, wp_ih1, wp_hh1, b_ih1, b_hh1, h2_seq);

    classify_kernel<<<dim3(B_SZ), dim3(H_SZ), 0, stream>>>(h2_seq, W_out, b_out, (float*)d_out);
}
